// VectorQuantizer_23974507446366
// MI455X (gfx1250) — matrix operand order repaired and checked by reading, not compiled, not run on hardware
//
#include <hip/hip_runtime.h>

typedef _Float16 half_t;
typedef __attribute__((ext_vector_type(8)))  _Float16 v8h;
typedef __attribute__((ext_vector_type(16))) _Float16 v16h;
typedef __attribute__((ext_vector_type(8)))  float    v8f;
typedef __attribute__((ext_vector_type(4)))  float    v4f;
typedef __attribute__((ext_vector_type(4)))  int      v4i;

// address-space-qualified 16-byte vector pointers for the async-LDS builtin
typedef __attribute__((address_space(1))) v4i as1_v4i;
typedef __attribute__((address_space(3))) v4i as3_v4i;

#define DIM      256
#define NCODES   8192
#define NROWS    32768        // B*S = 8*4096
#define MBLOCK   128          // rows per workgroup: 8 waves x 16
#define KT       32           // codebook tile (codes per iteration, 2 col-tiles/wave)
#define LDSW     264          // 256 + 8 halves pad -> conflict-free ds_load_b128

// gfx1250 async global->LDS path (ASYNCcnt), guarded: fallback = reg-staged copy
#if __has_builtin(__builtin_amdgcn_global_load_async_to_lds_b128)
#define USE_ASYNC_COPY 1
#else
#define USE_ASYNC_COPY 0
#endif
#if __has_builtin(__builtin_amdgcn_s_wait_asynccnt)
#define WAIT_ASYNC() __builtin_amdgcn_s_wait_asynccnt(0)
#else
#define WAIT_ASYNC() asm volatile("s_wait_asynccnt 0x0" ::: "memory")
#endif

// workspace layout (bytes)
#define WS_CB16   0                                   // 8192*256*2  = 4 MB
#define WS_IN16   (4u*1024u*1024u)                    // 32768*256*2 = 16 MB
#define WS_CNORM  (WS_IN16 + 16u*1024u*1024u)         // 8192*4
#define WS_IDX    (WS_CNORM + NCODES*4u)              // 32768*4
#define WS_PART   (WS_IDX + NROWS*4u)                 // 4096*4
#define NPART     4096

// d_out layout (floats), outputs concatenated in reference return order
#define OUT_QOUT  0u
#define OUT_QL    ((unsigned)NROWS*DIM)
#define OUT_EL    ((unsigned)NROWS*DIM + 1u)
#define OUT_Q     ((unsigned)NROWS*DIM + 2u)
#define OUT_IDX   (2u*(unsigned)NROWS*DIM + 2u)

// ---------------- codebook: fp32 -> f16 + exact fp32 ||e||^2 ----------------
__global__ __launch_bounds__(256) void vq_prep_codebook(
    const float* __restrict__ cb, half_t* __restrict__ cb16,
    float* __restrict__ cnorm)
{
  const int wave = threadIdx.x >> 5, lane = threadIdx.x & 31;
  const int row  = blockIdx.x * 8 + wave;
  const float* src = cb + (size_t)row * DIM + lane * 8;
  v4f f0 = *(const v4f*)src;
  v4f f1 = *(const v4f*)(src + 4);
  v8h h;
  float ss = 0.f;
#pragma unroll
  for (int i = 0; i < 4; ++i) { h[i]     = (half_t)f0[i]; ss = fmaf(f0[i], f0[i], ss); }
#pragma unroll
  for (int i = 0; i < 4; ++i) { h[4 + i] = (half_t)f1[i]; ss = fmaf(f1[i], f1[i], ss); }
  *(v8h*)(cb16 + (size_t)row * DIM + lane * 8) = h;
#pragma unroll
  for (int off = 16; off >= 1; off >>= 1) ss += __shfl_xor(ss, off, 32);
  if (lane == 0) cnorm[row] = ss;
}

// ---------------- inputs: fp32 -> f16 ----------------
__global__ __launch_bounds__(256) void vq_prep_inputs(
    const float* __restrict__ x, half_t* __restrict__ x16)
{
  const size_t base = ((size_t)blockIdx.x * 256 + threadIdx.x) * 8;
  v4f f0 = *(const v4f*)(x + base);
  v4f f1 = *(const v4f*)(x + base + 4);
  v8h h;
#pragma unroll
  for (int i = 0; i < 4; ++i) { h[i] = (half_t)f0[i]; h[4 + i] = (half_t)f1[i]; }
  *(v8h*)(x16 + base) = h;
}

// ---------------- fused distance-GEMM + running argmin (WMMA) ----------------
__global__ __launch_bounds__(256, 1) void vq_argmin(
    const half_t* __restrict__ x16, const half_t* __restrict__ cb16,
    const float* __restrict__ cnorm, int* __restrict__ idx)
{
  __shared__ half_t lcb[2][KT * LDSW];       // double-buffered 32 codes x 256 dims
  const int tid    = threadIdx.x;
  const int wave   = tid >> 5;
  const int lane   = tid & 31;
  const int lane16 = lane & 15;
  const int hs     = lane >> 4;              // half-wave select

  const int rowBase = blockIdx.x * MBLOCK + wave * 16;

  // Preload this wave's A fragments (16 rows x 256 dims, f16) into VGPRs.
  // ISA 16-bit A 16x32 layout: lanes 0-15 -> K 0..7 & 16..23, lanes 16-31 -> K 8..15 & 24..31.
  v16h a[8];
  {
    const half_t* ar = x16 + (size_t)(rowBase + lane16) * DIM + hs * 8;
#pragma unroll
    for (int f = 0; f < 8; ++f) {
      v8h lo = *(const v8h*)(ar + f * 32);
      v8h hi = *(const v8h*)(ar + f * 32 + 16);
#pragma unroll
      for (int i = 0; i < 8; ++i) { a[f][i] = lo[i]; a[f][8 + i] = hi[i]; }
    }
  }

  float bestd[8];
  int   bestk[8];
#pragma unroll
  for (int r = 0; r < 8; ++r) { bestd[r] = 3.0e38f; bestk[r] = 0; }

  // cooperative staging map: 256 threads cover 32 codes x 512 B (64 B / thread)
  const int cc = tid >> 3;            // code slot 0..31
  const int cs = (tid & 7) * 32;      // 32-half (64 B) segment within the code row

  // two independent WMMA accumulator chains = two 16-column tiles per wave
  auto compute_tile = [&](int buf, int kb) {
    v8f acc0 = {0.f, 0.f, 0.f, 0.f, 0.f, 0.f, 0.f, 0.f};
    v8f acc1 = {0.f, 0.f, 0.f, 0.f, 0.f, 0.f, 0.f, 0.f};
    const half_t* base0 = &lcb[buf][lane16 * LDSW + hs * 8];
    const half_t* base1 = base0 + 16 * LDSW;
#pragma unroll
    for (int f = 0; f < 8; ++f) {
      v8h b00 = *(const v8h*)(base0 + f * 32);
      v8h b01 = *(const v8h*)(base0 + f * 32 + 16);
      v8h b10 = *(const v8h*)(base1 + f * 32);
      v8h b11 = *(const v8h*)(base1 + f * 32 + 16);
      v16h bv0, bv1;
#pragma unroll
      for (int i = 0; i < 8; ++i) {
        bv0[i] = b00[i]; bv0[8 + i] = b01[i];
        bv1[i] = b10[i]; bv1[8 + i] = b11[i];
      }
      acc0 = __builtin_amdgcn_wmma_f32_16x16x32_f16(false, a[f], false, bv0,
                                                    (short)0, acc0, false, false);
      acc1 = __builtin_amdgcn_wmma_f32_16x16x32_f16(false, a[f], false, bv1,
                                                    (short)0, acc1, false, false);
    }
    // dist(row, code) = ||e||^2 - 2 x.e ; C layout: col = lane&15, VGPR r -> row r + 8*hs
    const float cn0 = cnorm[kb + lane16];
    const float cn1 = cnorm[kb + 16 + lane16];
    const int   c0  = kb + lane16, c1 = c0 + 16;
#pragma unroll
    for (int r = 0; r < 8; ++r) {
      float d0 = fmaf(-2.0f, acc0[r], cn0);
      float d1 = fmaf(-2.0f, acc1[r], cn1);
      if (d0 < bestd[r]) { bestd[r] = d0; bestk[r] = c0; }
      if (d1 < bestd[r]) { bestd[r] = d1; bestk[r] = c1; }
    }
  };

#if USE_ASYNC_COPY
  auto stage = [&](int kb, int buf) {
    const half_t* src = cb16 + (size_t)(kb + cc) * DIM + cs;
    half_t* dst = &lcb[buf][cc * LDSW + cs];
    as1_v4i* g = (as1_v4i*)(unsigned long long)src;                 // AS1 int4*
    as3_v4i* l = (as3_v4i*)(unsigned)(unsigned long long)dst;       // AS3 int4*
#pragma unroll
    for (int i = 0; i < 4; ++i)
      __builtin_amdgcn_global_load_async_to_lds_b128(g + i, l + i, 0, 0);
  };
  stage(0, 0);
  WAIT_ASYNC();
  __syncthreads();
  for (int kb = 0; kb < NCODES; kb += KT) {
    const int cur = (kb >> 5) & 1;
    if (kb + KT < NCODES) stage(kb + KT, cur ^ 1);   // overlaps with WMMA below
    compute_tile(cur, kb);
    WAIT_ASYNC();          // own tile-(t+1) writes done
    __syncthreads();       // everyone's writes done, everyone's reads of cur done
  }
#else
  {   // prologue: tile 0 through registers
    const half_t* s = cb16 + (size_t)cc * DIM + cs;
    v8h r0 = ((const v8h*)s)[0], r1 = ((const v8h*)s)[1],
        r2 = ((const v8h*)s)[2], r3 = ((const v8h*)s)[3];
    half_t* d = &lcb[0][cc * LDSW + cs];
    ((v8h*)d)[0] = r0; ((v8h*)d)[1] = r1; ((v8h*)d)[2] = r2; ((v8h*)d)[3] = r3;
  }
  __syncthreads();
  for (int kb = 0; kb < NCODES; kb += KT) {
    const int  cur  = (kb >> 5) & 1;
    const bool more = (kb + KT < NCODES);
    v8h r0, r1, r2, r3;
    if (more) {   // issue next tile's global loads before the WMMA block
      const half_t* s = cb16 + (size_t)(kb + KT + cc) * DIM + cs;
      r0 = ((const v8h*)s)[0]; r1 = ((const v8h*)s)[1];
      r2 = ((const v8h*)s)[2]; r3 = ((const v8h*)s)[3];
    }
    compute_tile(cur, kb);
    if (more) {
      half_t* d = &lcb[cur ^ 1][cc * LDSW + cs];
      ((v8h*)d)[0] = r0; ((v8h*)d)[1] = r1; ((v8h*)d)[2] = r2; ((v8h*)d)[3] = r3;
    }
    __syncthreads();
  }
#endif

  // Reduce across the 16 lanes sharing each row set; tie-break to lowest index.
#pragma unroll
  for (int m = 8; m >= 1; m >>= 1) {
#pragma unroll
    for (int r = 0; r < 8; ++r) {
      float od = __shfl_xor(bestd[r], m, 32);
      int   ok = __shfl_xor(bestk[r], m, 32);
      if (od < bestd[r] || (od == bestd[r] && ok < bestk[r])) {
        bestd[r] = od; bestk[r] = ok;
      }
    }
  }
  if (lane16 == 0) {
#pragma unroll
    for (int r = 0; r < 8; ++r)
      idx[rowBase + hs * 8 + r] = bestk[r];
  }
}

// ---------------- gather + partial MSE (deterministic, no atomics) ----------------
__global__ __launch_bounds__(256) void vq_gather(
    const float* __restrict__ x, const float* __restrict__ cb,
    const int* __restrict__ idx, float* __restrict__ out,
    float* __restrict__ part)
{
  __shared__ float wsum[8];
  const size_t gid = (size_t)blockIdx.x * 256 + threadIdx.x;
  const size_t e0  = gid * 8;
  const int row = (int)(e0 >> 8);
  const int d0  = (int)(e0 & 255);
  const int k   = idx[row];

  const float* qp = cb + (size_t)k * DIM + d0;
  v4f q0 = *(const v4f*)qp, q1 = *(const v4f*)(qp + 4);
  const float* xp = x + e0;
  v4f x0 = *(const v4f*)xp, x1 = *(const v4f*)(xp + 4);

  // quantized_out region (16B aligned): forward value == quantized
  *(v4f*)(out + OUT_QOUT + e0)     = q0;
  *(v4f*)(out + OUT_QOUT + e0 + 4) = q1;

  // quantized region starts at +2 floats -> scalar stores
  float s = 0.f;
#pragma unroll
  for (int i = 0; i < 4; ++i) {
    out[OUT_Q + e0 + i] = q0[i];
    float d = q0[i] - x0[i]; s = fmaf(d, d, s);
  }
#pragma unroll
  for (int i = 0; i < 4; ++i) {
    out[OUT_Q + e0 + 4 + i] = q1[i];
    float d = q1[i] - x1[i]; s = fmaf(d, d, s);
  }
  if (d0 == 0) out[OUT_IDX + row] = (float)k;

#pragma unroll
  for (int off = 16; off >= 1; off >>= 1) s += __shfl_xor(s, off, 32);
  if ((threadIdx.x & 31) == 0) wsum[threadIdx.x >> 5] = s;
  __syncthreads();
  if (threadIdx.x == 0) {
    float t = 0.f;
    for (int w = 0; w < 8; ++w) t += wsum[w];
    part[blockIdx.x] = t;
  }
}

// ---------------- final loss reduction ----------------
__global__ __launch_bounds__(256) void vq_loss(
    const float* __restrict__ part, float* __restrict__ out)
{
  __shared__ float sh[256];
  float s = 0.f;
  for (int i = threadIdx.x; i < NPART; i += 256) s += part[i];
  sh[threadIdx.x] = s;
  __syncthreads();
  for (int off = 128; off >= 1; off >>= 1) {
    if (threadIdx.x < off) sh[threadIdx.x] += sh[threadIdx.x + off];
    __syncthreads();
  }
  if (threadIdx.x == 0) {
    float L = sh[0] / (float)(NROWS * DIM);
    out[OUT_QL] = L;   // q_latent_loss (forward value)
    out[OUT_EL] = L;   // e_latent_loss (same forward value)
  }
}

extern "C" void kernel_launch(void* const* d_in, const int* in_sizes, int n_in,
                              void* d_out, int out_size, void* d_ws, size_t ws_size,
                              hipStream_t stream) {
  const float* x  = (const float*)d_in[0];   // inputs   [8,4096,256] fp32
  const float* cb = (const float*)d_in[1];   // codebook [8192,256]   fp32
  float* out = (float*)d_out;

  char* ws = (char*)d_ws;
  half_t* cb16  = (half_t*)(ws + WS_CB16);
  half_t* x16   = (half_t*)(ws + WS_IN16);
  float*  cnorm = (float*)(ws + WS_CNORM);
  int*    idx   = (int*)(ws + WS_IDX);
  float*  part  = (float*)(ws + WS_PART);

  vq_prep_codebook<<<NCODES / 8, 256, 0, stream>>>(cb, cb16, cnorm);
  vq_prep_inputs<<<(NROWS * DIM) / (256 * 8), 256, 0, stream>>>(x, x16);
  vq_argmin<<<NROWS / MBLOCK, 256, 0, stream>>>(x16, cb16, cnorm, idx);
  vq_gather<<<(NROWS * DIM) / (256 * 8), 256, 0, stream>>>(x, cb, idx, out, part);
  vq_loss<<<1, 256, 0, stream>>>(part, out);
}
